// Modified_GCN_GRU_72095321031265
// MI455X (gfx1250) — matrix operand order repaired and checked by reading, not compiled, not run on hardware
//
#include <hip/hip_runtime.h>
#include <hip/hip_bf16.h>

// ---------------------------------------------------------------------------
// Types for CDNA5 WMMA (wave32): 16x16x32 bf16 -> f32 accumulate
// ---------------------------------------------------------------------------
typedef __attribute__((ext_vector_type(16))) __bf16 v16bf;
typedef __attribute__((ext_vector_type(8)))  __bf16 v8bf;
typedef __attribute__((ext_vector_type(8)))  float  v8f;

#define GCN_EPS 1e-9f

__device__ __forceinline__ __bf16 f32_to_bf16(float f) {
  unsigned u = __float_as_uint(f);
  unsigned r = (u + 0x7FFFu + ((u >> 16) & 1u)) >> 16;  // RNE
  unsigned short h = (unsigned short)r;
  return __builtin_bit_cast(__bf16, h);
}

__device__ __forceinline__ float bf16_to_f32(__bf16 b) {
  unsigned short h = __builtin_bit_cast(unsigned short, b);
  return __uint_as_float(((unsigned)h) << 16);
}

// ---------------------------------------------------------------------------
// Utility: zero float buffer
// ---------------------------------------------------------------------------
__global__ void zero_f32_kernel(float* __restrict__ p, long long n) {
  long long i = (long long)blockIdx.x * blockDim.x + threadIdx.x;
  if (i < n) p[i] = 0.0f;
}

// ---------------------------------------------------------------------------
// Degree accumulation over edges (int64 edge index)
// ---------------------------------------------------------------------------
__global__ void degree_kernel(const long long* __restrict__ ei, long long E,
                              float* __restrict__ degA,   // out-degree (src)
                              float* __restrict__ degB) { // in-degree (dst)
  long long i = (long long)blockIdx.x * blockDim.x + threadIdx.x;
  if (i >= E) return;
  long long s = ei[i];
  long long d = ei[E + i];
  atomicAdd(&degA[s], 1.0f);
  atomicAdd(&degB[d], 1.0f);
}

// cnode[n] = rsqrt(indeg+1) * 1/(sqrt(deg_tot+eps)+eps)
__global__ void node_norm_kernel(const float* __restrict__ degA,
                                 const float* __restrict__ degB,
                                 float* __restrict__ cnode, int N) {
  int n = blockIdx.x * blockDim.x + threadIdx.x;
  if (n >= N) return;
  float dt = degA[n] + degB[n];
  float dinv_ext = 1.0f / (sqrtf(dt + GCN_EPS) + GCN_EPS);
  float dinv = rsqrtf(degB[n] + 1.0f);
  cnode[n] = dinv * dinv_ext;
}

// ---------------------------------------------------------------------------
// f32 -> bf16 elementwise
// ---------------------------------------------------------------------------
__global__ void cvt_bf16_kernel(const float* __restrict__ in,
                                __bf16* __restrict__ out, long long n) {
  long long i = (long long)blockIdx.x * blockDim.x + threadIdx.x;
  if (i < n) out[i] = f32_to_bf16(in[i]);
}

// 128x128 transpose + convert: out[n*128+k] = bf16(in[k*128+n])
__global__ void transpose_cvt128_kernel(const float* __restrict__ in,
                                        __bf16* __restrict__ out) {
  int n = blockIdx.x;   // 0..127
  int k = threadIdx.x;  // 0..127
  out[n * 128 + k] = f32_to_bf16(in[k * 128 + n]);
}

// ---------------------------------------------------------------------------
// WMMA GEMM, K fixed = 128:  C[M,N] = A[M,128] * B[128,N]
//   A   : bf16 row-major [M,128]
//   Bt  : bf16 [N,128]  (column-major view of B, i.e. Bt[n][k] = B[k][n])
//   C   : f32 row-major [M,N]
// One 16x32 output tile per wave (two accumulators share one A fragment);
// 8 waves per 256-thread block. All fragments for the K=128 reduction are
// preloaded (24 x b128 loads) so the 8 v_wmma issue back-to-back.
//
// A fragment (16x32 bf16): lane<16 holds row M=lane, K = k0+0..15;
// lane>=16 holds row M=lane-16, K = k0+16..31.
// B fragment (32x16 bf16): lane<16 holds col N=lane, K = k0+0..15;
// lane>=16 holds col N=lane-16, K = k0+16..31.
// ---------------------------------------------------------------------------
__global__ __launch_bounds__(256) void wmma_gemm_k128_bf16(
    const __bf16* __restrict__ A, const __bf16* __restrict__ Bt,
    float* __restrict__ C, int M, int N) {
  int wave = (blockIdx.x << 3) + (threadIdx.x >> 5);
  int lane = threadIdx.x & 31;
  int ntiles = N >> 5;                 // 32-wide N tiles per wave
  int tiles = (M >> 4) * ntiles;
  if (wave >= tiles) return;  // wave-uniform: EXEC stays all-ones for WMMA
  int mt = wave / ntiles;
  int nt = wave - mt * ntiles;
  int l15 = lane & 15;
  int hi = lane >> 4;

  const __bf16* arow  = A + (long long)(mt * 16 + l15) * 128;
  const __bf16* brow0 = Bt + (long long)(nt * 32 + l15) * 128;
  const __bf16* brow1 = Bt + (long long)(nt * 32 + 16 + l15) * 128;

  v16bf a[4], b0[4], b1[4];
#pragma unroll
  for (int ks = 0; ks < 4; ++ks) {
    int k0 = ks * 32;
    a[ks] = *(const v16bf*)(arow + k0 + 16 * hi);
    b0[ks] = *(const v16bf*)(brow0 + k0 + 16 * hi);
    b1[ks] = *(const v16bf*)(brow1 + k0 + 16 * hi);
  }

  v8f acc0 = {};
  v8f acc1 = {};
#pragma unroll
  for (int ks = 0; ks < 4; ++ks) {
    acc0 = __builtin_amdgcn_wmma_f32_16x16x32_bf16(
        false, a[ks], false, b0[ks], (short)0, acc0, false, false);
    acc1 = __builtin_amdgcn_wmma_f32_16x16x32_bf16(
        false, a[ks], false, b1[ks], (short)0, acc1, false, false);
  }

  // C layout: VGPR v -> row mt*16 + hi*8 + v; cols nt*32 + {0,16} + l15
  float* crow0 = C + (long long)(mt * 16 + hi * 8) * N + nt * 32 + l15;
  float* crow1 = crow0 + 16;
#pragma unroll
  for (int v = 0; v < 8; ++v) {
    crow0[(long long)v * N] = acc0[v];
    crow1[(long long)v * N] = acc1[v];
  }
}

// ---------------------------------------------------------------------------
// Edge aggregation: one wave per edge, 32 lanes x float4 = 128 floats
// agg[dst] += xw[src] * cnode[src]*cnode[dst]
// ---------------------------------------------------------------------------
__global__ __launch_bounds__(256) void edge_agg_kernel(
    const long long* __restrict__ ei, long long E,
    const float* __restrict__ cnode,
    const float4* __restrict__ xw4,   // [N*32]
    float* __restrict__ agg) {        // [N*128]
  long long e = ((long long)blockIdx.x << 3) + (threadIdx.x >> 5);
  if (e >= E) return;
  int lane = threadIdx.x & 31;
  long long s = ei[e];
  long long d = ei[E + e];
  float coeff = cnode[s] * cnode[d];
  float4 v = xw4[s * 32 + lane];
  float* dst = agg + d * 128 + lane * 4;
  atomicAdd(dst + 0, v.x * coeff);
  atomicAdd(dst + 1, v.y * coeff);
  atomicAdd(dst + 2, v.z * coeff);
  atomicAdd(dst + 3, v.w * coeff);
}

// ---------------------------------------------------------------------------
// Epilogue 1: h_neigh = relu(agg + cnode^2 * xw + conv_b) -> bf16
// ---------------------------------------------------------------------------
__global__ void epilogue1_kernel(const float* __restrict__ agg,
                                 const float* __restrict__ xw,
                                 const float* __restrict__ cnode,
                                 const float* __restrict__ conv_b,
                                 __bf16* __restrict__ hn_bf, long long total) {
  long long i = (long long)blockIdx.x * blockDim.x + threadIdx.x;
  if (i >= total) return;
  int n = (int)(i >> 7);
  int d = (int)(i & 127);
  float c = cnode[n];
  c = c * c;
  float v = agg[i] + c * xw[i] + conv_b[d];
  v = fmaxf(v, 0.0f);
  hn_bf[i] = f32_to_bf16(v);
}

// ---------------------------------------------------------------------------
// Epilogue 2: h = sigmoid(h_neigh + h_self + hnWt - xWt) -> bf16
// ---------------------------------------------------------------------------
__global__ void epilogue2_kernel(const __bf16* __restrict__ hn_bf,
                                 const float* __restrict__ hself,
                                 const float* __restrict__ hnwt,
                                 const float* __restrict__ xwt,
                                 __bf16* __restrict__ h_bf, long long total) {
  long long i = (long long)blockIdx.x * blockDim.x + threadIdx.x;
  if (i >= total) return;
  float v = bf16_to_f32(hn_bf[i]) + hself[i] + hnwt[i] - xwt[i];
  float s = 1.0f / (1.0f + __expf(-v));
  h_bf[i] = f32_to_bf16(s);
}

// ---------------------------------------------------------------------------
// Sequential GRU scan: one persistent workgroup, 768 threads (24 waves).
// Thread j computes gh[j] = W_hh[j,:] . h + b_hh[j] (256 fp32 FMAs, W_hh is
// L2-resident). Threads 0..255 then apply the gate math. h and gh staged in
// LDS; two barriers per step. xp[t] is loaded before the recurrent matvec and
// xp[t+1] is prefetched (global_prefetch_b8) to overlap the next step's input
// stream with compute.
// ---------------------------------------------------------------------------
__global__ __launch_bounds__(768) void gru_scan_kernel(
    const float* __restrict__ xp,     // [N,768]
    const float* __restrict__ Whh,    // [768,256]
    const float* __restrict__ b_ih,   // [768]
    const float* __restrict__ b_hh,   // [768]
    float* __restrict__ out,          // [N,256]
    int N) {
  __shared__ float sh[256];
  __shared__ float sg[768];
  int j = threadIdx.x;
  if (j < 256) sh[j] = 0.0f;
  float bhh = b_hh[j];
  float bi0 = 0.f, bi1 = 0.f, bi2 = 0.f;
  if (j < 256) {
    bi0 = b_ih[j];
    bi1 = b_ih[256 + j];
    bi2 = b_ih[512 + j];
  }
  const float4* w4 = (const float4*)(Whh + (long long)j * 256);
  __syncthreads();

  for (int t = 0; t < N; ++t) {
    // Input-projection row for this step: independent of h, load early.
    const float* xr = xp + (long long)t * 768;
    float x0 = 0.f, x1 = 0.f, x2 = 0.f;
    if (j < 256) {
      x0 = xr[j];
      x1 = xr[256 + j];
      x2 = xr[512 + j];
    }
    // Prefetch next step's row while the recurrent matvec runs.
    if (t + 1 < N) __builtin_prefetch(xr + 768 + j, 0, 0);

    float acc = bhh;
    const float4* h4 = (const float4*)sh;  // LDS broadcast reads
#pragma unroll 8
    for (int k = 0; k < 64; ++k) {
      float4 w = w4[k];
      float4 h = h4[k];
      acc += w.x * h.x + w.y * h.y + w.z * h.z + w.w * h.w;
    }
    sg[j] = acc;
    __syncthreads();
    if (j < 256) {
      float r = 1.0f / (1.0f + __expf(-(x0 + bi0 + sg[j])));
      float z = 1.0f / (1.0f + __expf(-(x1 + bi1 + sg[256 + j])));
      float n = tanhf(x2 + bi2 + r * sg[512 + j]);
      float hn = (1.0f - z) * n + z * sh[j];
      out[(long long)t * 256 + j] = hn;
      sh[j] = hn;
    }
    __syncthreads();
  }
}

// ---------------------------------------------------------------------------
// Host-side orchestration
// ---------------------------------------------------------------------------
extern "C" void kernel_launch(void* const* d_in, const int* in_sizes, int n_in,
                              void* d_out, int out_size, void* d_ws, size_t ws_size,
                              hipStream_t stream) {
  const float*     x      = (const float*)d_in[0];
  const long long* ei     = (const long long*)d_in[1];
  const float*     conv_W = (const float*)d_in[2];
  const float*     conv_b = (const float*)d_in[3];
  const float*     W_0    = (const float*)d_in[4];
  const float*     W_t    = (const float*)d_in[5];
  const float*     W_ih   = (const float*)d_in[6];
  const float*     W_hh   = (const float*)d_in[7];
  const float*     b_ih   = (const float*)d_in[8];
  const float*     b_hh   = (const float*)d_in[9];
  float* out = (float*)d_out;

  const long long N = in_sizes[0] / 128;  // 20000
  const long long E = in_sizes[1] / 2;    // 640000
  const long long ND = N * 128;

  // ---- carve workspace -----------------------------------------------------
  char* ws = (char*)d_ws;
  size_t off = 0;
  auto carve = [&](size_t bytes) -> char* {
    char* p = ws + off;
    off = (off + bytes + 255) & ~(size_t)255;
    return p;
  };
  float*  degA  = (float*)carve(N * 4);
  float*  degB  = (float*)carve(N * 4);
  float*  cnode = (float*)carve(N * 4);
  __bf16* wcT   = (__bf16*)carve(128 * 128 * 2);
  __bf16* w0T   = (__bf16*)carve(128 * 128 * 2);
  __bf16* wtT   = (__bf16*)carve(128 * 128 * 2);
  __bf16* wihB  = (__bf16*)carve(768 * 128 * 2);  // W_ih [768,128] == Bt layout
  __bf16* xbf   = (__bf16*)carve(ND * 2);
  float*  xw    = (float*)carve(ND * 4);
  float*  agg   = (float*)carve(ND * 4);
  float*  hself = (float*)carve(ND * 4);
  float*  xwt   = (float*)carve(ND * 4);
  __bf16* hnbf  = (__bf16*)carve(ND * 2);
  __bf16* hbf   = (__bf16*)carve(ND * 2);
  float*  xp    = (float*)carve(N * 768 * 4);
  float*  hnwt  = xw;  // alias: xw is dead after epilogue1

  // ---- 1) zero accumulators ------------------------------------------------
  zero_f32_kernel<<<(int)((N + 255) / 256), 256, 0, stream>>>(degA, N);
  zero_f32_kernel<<<(int)((N + 255) / 256), 256, 0, stream>>>(degB, N);
  zero_f32_kernel<<<(int)((ND + 255) / 256), 256, 0, stream>>>(agg, ND);

  // ---- 2) degrees + per-node coefficient ------------------------------------
  degree_kernel<<<(int)((E + 255) / 256), 256, 0, stream>>>(ei, E, degA, degB);
  node_norm_kernel<<<(int)((N + 255) / 256), 256, 0, stream>>>(degA, degB, cnode, (int)N);

  // ---- 3) bf16 conversions --------------------------------------------------
  cvt_bf16_kernel<<<(int)((ND + 255) / 256), 256, 0, stream>>>(x, xbf, ND);
  cvt_bf16_kernel<<<(768 * 128 + 255) / 256, 256, 0, stream>>>(W_ih, wihB, 768 * 128);
  transpose_cvt128_kernel<<<128, 128, 0, stream>>>(conv_W, wcT);
  transpose_cvt128_kernel<<<128, 128, 0, stream>>>(W_0, w0T);
  transpose_cvt128_kernel<<<128, 128, 0, stream>>>(W_t, wtT);

  // ---- 4) WMMA GEMMs from x -------------------------------------------------
  {
    int tiles = (int)(N / 16) * (128 / 32);
    int blocks = (tiles + 7) / 8;
    wmma_gemm_k128_bf16<<<blocks, 256, 0, stream>>>(xbf, wcT, xw, (int)N, 128);
    wmma_gemm_k128_bf16<<<blocks, 256, 0, stream>>>(xbf, w0T, hself, (int)N, 128);
    wmma_gemm_k128_bf16<<<blocks, 256, 0, stream>>>(xbf, wtT, xwt, (int)N, 128);
  }

  // ---- 5) edge aggregation (one wave per edge) ------------------------------
  edge_agg_kernel<<<(int)((E + 7) / 8), 256, 0, stream>>>(
      ei, E, cnode, (const float4*)xw, agg);

  // ---- 6) epilogue 1: h_neigh -----------------------------------------------
  epilogue1_kernel<<<(int)((ND + 255) / 256), 256, 0, stream>>>(
      agg, xw, cnode, conv_b, hnbf, ND);

  // ---- 7) h_neigh @ W_t (reuses xw buffer) ----------------------------------
  {
    int tiles = (int)(N / 16) * (128 / 32);
    int blocks = (tiles + 7) / 8;
    wmma_gemm_k128_bf16<<<blocks, 256, 0, stream>>>(hnbf, wtT, hnwt, (int)N, 128);
  }

  // ---- 8) epilogue 2: h = sigmoid(...) --------------------------------------
  epilogue2_kernel<<<(int)((ND + 255) / 256), 256, 0, stream>>>(
      hnbf, hself, hnwt, xwt, hbf, ND);

  // ---- 9) input projections xp = h @ W_ih^T ---------------------------------
  {
    int tiles = (int)(N / 16) * (768 / 32);
    int blocks = (tiles + 7) / 8;
    wmma_gemm_k128_bf16<<<blocks, 256, 0, stream>>>(hbf, wihB, xp, (int)N, 768);
  }

  // ---- 10) sequential GRU scan (single persistent workgroup) ----------------
  gru_scan_kernel<<<1, 768, 0, stream>>>(xp, W_hh, b_ih, b_hh, out, (int)N);
}
